// UDF_40578851012637
// MI455X (gfx1250) — matrix operand order repaired and checked by reading, not compiled, not run on hardware
//
#include <hip/hip_runtime.h>

// ---------------------------------------------------------------------------
// Types for CDNA5 WMMA
// ---------------------------------------------------------------------------
typedef _Float16 half_t;
typedef __attribute__((ext_vector_type(16))) _Float16 v16h;
typedef __attribute__((ext_vector_type(8)))  float    v8f;

#define BN_EPS 1e-5f
#define SLOPE  0.2f

__device__ __forceinline__ void lds_fence() {
    asm volatile("s_wait_dscnt 0x0" ::: "memory");
}

// CDNA5 async memory->LDS copy (ASYNCcnt-tracked). lds_off is the byte offset
// inside the wave's LDS allocation (dynamic LDS starts at 0 here: no static
// __shared__ in the MLP kernel).
__device__ __forceinline__ void async_copy_b128(unsigned lds_off, const void* gaddr) {
    asm volatile("global_load_async_to_lds_b128 %0, %1, off"
                 :: "v"(lds_off), "v"(gaddr)
                 : "memory");
}
__device__ __forceinline__ void wait_async() {
    asm volatile("s_wait_asynccnt 0x0" ::: "memory");
}

constexpr int cmax2(int a, int b) { return a > b ? a : b; }
constexpr int cmax5(int a, int b, int c, int d, int e) {
    return cmax2(a, cmax2(b, cmax2(c, cmax2(d, e))));
}

// ---------------------------------------------------------------------------
// KNN: one wave32 per query point. Per-lane register top-10 then LDS merge.
// ---------------------------------------------------------------------------
__global__ __launch_bounds__(256) void knn_kernel(
    const float* __restrict__ xyz,    // [B][N][3]
    const float* __restrict__ query,  // [B][3][M]
    int* __restrict__ idx,            // [B][M][10]
    int B, int N, int M)
{
    const int lane = threadIdx.x & 31;
    const int wave = threadIdx.x >> 5;
    const int qid  = blockIdx.x * 8 + wave;
    if (qid >= B * M) return;
    const int b = qid / M, m = qid - b * M;

    const float qx = query[(size_t)b * 3 * M + 0 * M + m];
    const float qy = query[(size_t)b * 3 * M + 1 * M + m];
    const float qz = query[(size_t)b * 3 * M + 2 * M + m];

    float bd[10]; int bi[10];
#pragma unroll
    for (int j = 0; j < 10; ++j) { bd[j] = 3.0e38f; bi[j] = 0; }

    const float* base = xyz + (size_t)b * N * 3;
    for (int n = lane; n < N; n += 32) {
        float dx = qx - base[n * 3 + 0];
        float dy = qy - base[n * 3 + 1];
        float dz = qz - base[n * 3 + 2];
        float d2 = dx * dx + dy * dy + dz * dz;
        if (d2 < bd[9]) {
            float cd = d2; int ci = n;
#pragma unroll
            for (int j = 0; j < 10; ++j) {
                if (cd < bd[j]) {
                    float td = bd[j]; bd[j] = cd; cd = td;
                    int   ti = bi[j]; bi[j] = ci; ci = ti;
                }
            }
        }
    }

    __shared__ float sdl[8][320];
    __shared__ int   sil[8][320];
#pragma unroll
    for (int j = 0; j < 10; ++j) {
        sdl[wave][lane * 10 + j] = bd[j];
        sil[wave][lane * 10 + j] = bi[j];
    }
    __syncthreads();

    if (lane == 0) {
        for (int l = 1; l < 32; ++l) {
#pragma unroll
            for (int j = 0; j < 10; ++j) {
                float cd = sdl[wave][l * 10 + j];
                int   ci = sil[wave][l * 10 + j];
                if (cd < bd[9]) {
#pragma unroll
                    for (int jj = 0; jj < 10; ++jj) {
                        if (cd < bd[jj]) {
                            float td = bd[jj]; bd[jj] = cd; cd = td;
                            int   ti = bi[jj]; bi[jj] = ci; ci = ti;
                        }
                    }
                }
            }
        }
        int* op = idx + (size_t)qid * 10;
#pragma unroll
        for (int j = 0; j < 10; ++j) op[j] = bi[j];
    }
}

// ---------------------------------------------------------------------------
// cv1 assembly: local offset, oriented normal, |signed dist|, f16 padded rows
// ---------------------------------------------------------------------------
__global__ void cv1_kernel(
    const float* __restrict__ xyz, const float* __restrict__ nrm,
    const float* __restrict__ query, const int* __restrict__ idx,
    half_t* __restrict__ cv1h,    // [rows][32] zero padded
    float* __restrict__ dist,     // [rows]
    float* __restrict__ onrm,     // [rows][3]
    int B, int N, int M, int K)
{
    int t = blockIdx.x * blockDim.x + threadIdx.x;
    int total = B * M * K;
    if (t >= total) return;
    int bm = t / K;
    int m  = bm % M;
    int b  = bm / M;

    float qx = query[(size_t)b * 3 * M + 0 * M + m];
    float qy = query[(size_t)b * 3 * M + 1 * M + m];
    float qz = query[(size_t)b * 3 * M + 2 * M + m];

    int n = idx[t];
    const float* p  = xyz + ((size_t)b * N + n) * 3;
    const float* nr = nrm + ((size_t)b * N + n) * 3;
    float lx = qx - p[0], ly = qy - p[1], lz = qz - p[2];
    float nx = nr[0], ny = nr[1], nz = nr[2];
    float sd = lx * nx + ly * ny + lz * nz;
    float s  = (sd > 0.f) ? 1.f : ((sd < 0.f) ? -1.f : 0.f);
    float ox = s * nx, oy = s * ny, oz = s * nz;

    dist[t] = fabsf(sd);
    onrm[(size_t)t * 3 + 0] = ox;
    onrm[(size_t)t * 3 + 1] = oy;
    onrm[(size_t)t * 3 + 2] = oz;

    half_t* c = cv1h + (size_t)t * 32;
    c[0] = (half_t)lx; c[1] = (half_t)ly; c[2] = (half_t)lz;
    c[3] = (half_t)ox; c[4] = (half_t)oy; c[5] = (half_t)oz;
#pragma unroll
    for (int j = 6; j < 32; ++j) c[j] = (half_t)0.f;
}

// ---------------------------------------------------------------------------
// Weight packing: f32 W[Cout][Cin] -> f16 WMMA B-fragments.
// Fragment lane layout (32x16 KxN): lane = (khalf<<4)|n, halves h: K = khalf*16+h
// ---------------------------------------------------------------------------
__global__ void pack_wb(const float* __restrict__ W, half_t* __restrict__ dst,
                        int Cout, int Cin, int Kp, int Np)
{
    int T = (Np / 16) * (Kp / 32) * 512;
    int t = blockIdx.x * blockDim.x + threadIdx.x;
    if (t >= T) return;
    int h    = t & 15;
    int lane = (t >> 4) & 31;
    int frag = t >> 9;
    int kt   = frag % (Kp / 32);
    int nt   = frag / (Kp / 32);
    int n = nt * 16 + (lane & 15);
    int k = kt * 32 + (lane >> 4) * 16 + h;
    float v = (n < Cout && k < Cin) ? W[(size_t)n * Cin + k] : 0.f;
    dst[t] = (half_t)v;
}

// Fold bias + BN into per-channel scale/shift: y = acc*sc[c] + sh[c]
__global__ void pack_ss(const float* __restrict__ b, const float* __restrict__ g,
                        const float* __restrict__ beta, const float* __restrict__ mu,
                        const float* __restrict__ var,
                        float* __restrict__ ss, int Cout, int Np, int last)
{
    int c = threadIdx.x;
    if (c >= Np) return;
    float sc = 0.f, sh = 0.f;
    if (c < Cout) {
        if (last) { sc = 1.f; sh = b[c]; }
        else {
            float s = g[c] * rsqrtf(var[c] + BN_EPS);
            sc = s;
            sh = (b[c] - mu[c]) * s + beta[c];
        }
    }
    ss[c] = sc;
    ss[Np + c] = sh;
}

// ---------------------------------------------------------------------------
// One WMMA GEMM layer over R 16-row LDS tiles owned by one wave.
// Each B-fragment load is amortized over R WMMAs (weight-bandwidth fix).
// OUTMODE: 0 = LDS f16 + LeakyReLU, 1 = global f16 (no act), 2 = global f32 ch0
// ---------------------------------------------------------------------------
template<int Cin, int Cout, int R, int OUTMODE>
__device__ __forceinline__ void run_layer(
    const half_t* __restrict__ actIn,   // LDS [16R][Cin]
    half_t* actOut,                     // LDS [16R][Cout]
    const half_t* __restrict__ wf,      // packed B fragments
    const float* __restrict__ ss,       // [2][Cout]
    half_t* __restrict__ gOut,          // OUTMODE 1
    float* __restrict__ sOut,           // OUTMODE 2
    int lane)
{
    const int m  = lane & 15;
    const int kh = lane >> 4;
    for (int nt = 0; nt < Cout / 16; ++nt) {
        v8f zero = {};
        v8f acc[R];
#pragma unroll
        for (int r = 0; r < R; ++r) acc[r] = zero;
#pragma unroll
        for (int kt = 0; kt < Cin / 32; ++kt) {
            union UV { v16h v; uint4 q[2]; } Bf;
            const half_t* wp = wf + ((size_t)(nt * (Cin / 32) + kt) * 32 + lane) * 16;
            Bf.q[0] = ((const uint4*)wp)[0];
            Bf.q[1] = ((const uint4*)wp)[1];
#pragma unroll
            for (int r = 0; r < R; ++r) {
                union UV A;
                const half_t* ar = actIn + (r * 16 + m) * Cin + kt * 32;
                A.q[0] = *(const uint4*)(ar + kh * 16);
                A.q[1] = *(const uint4*)(ar + kh * 16 + 8);
                acc[r] = __builtin_amdgcn_wmma_f32_16x16x32_f16(
                    false, A.v, false, Bf.v, (short)0, acc[r], false, false);
            }
        }
        const int n = nt * 16 + m;
        const float s = ss[n];
        const float t = ss[Cout + n];
#pragma unroll
        for (int r = 0; r < R; ++r) {
#pragma unroll
            for (int v = 0; v < 8; ++v) {
                float y = acc[r][v] * s + t;
                const int mr = r * 16 + v + 8 * kh;   // C/D row = vgpr + 8*(lane>>4)
                if (OUTMODE == 0) {
                    y = (y >= 0.f) ? y : SLOPE * y;
                    actOut[mr * Cout + n] = (half_t)y;
                } else if (OUTMODE == 1) {
                    gOut[(size_t)mr * Cout + n] = (half_t)y;
                } else {
                    if (m == 0) sOut[mr] = y;   // channel 0 only
                }
            }
        }
    }
}

// ---------------------------------------------------------------------------
// 4-layer MLP: one wave per block, R row-tiles per wave, LDS double buffer.
// Input staged via CDNA5 async memory->LDS copies.
// ---------------------------------------------------------------------------
template<int C0, int C1, int C2, int C3, int C4, bool PATCH, int R>
__global__ __launch_bounds__(32) void mlp4_kernel(
    const half_t* __restrict__ in, int rows,
    const half_t* __restrict__ w0, const float* __restrict__ ss0,
    const half_t* __restrict__ w1, const float* __restrict__ ss1,
    const half_t* __restrict__ w2, const float* __restrict__ ss2,
    const half_t* __restrict__ w3, const float* __restrict__ ss3,
    half_t* __restrict__ featOut, float* __restrict__ scalarOut)
{
    constexpr int CMAX = cmax5(C0, C1, C2, C3, C4);
    extern __shared__ half_t smem[];
    const int lane = threadIdx.x & 31;
    const int rowBase = blockIdx.x * (16 * R);
    if (rowBase >= rows) return;

    half_t* bufA = smem;                       // LDS byte offset 0
    half_t* bufB = smem + 16 * R * CMAX;

    // Stage input rows straight into LDS with async 128-bit copies
    {
        const uint4* src = (const uint4*)(in + (size_t)rowBase * C0);
        constexpr int NQ = 16 * R * C0 / 8;    // uint4 count, multiple of 32
        for (int i = lane; i < NQ; i += 32)
            async_copy_b128((unsigned)(i * 16), (const void*)(src + i));
        wait_async();
        lds_fence();
    }

    run_layer<C0, C1, R, 0>(bufA, bufB, w0, ss0, nullptr, nullptr, lane);
    lds_fence();
    run_layer<C1, C2, R, 0>(bufB, bufA, w1, ss1, nullptr, nullptr, lane);
    lds_fence();
    run_layer<C2, C3, R, 0>(bufA, bufB, w2, ss2, nullptr, nullptr, lane);
    lds_fence();
    if (PATCH) {
        run_layer<C3, C4, R, 1>(bufB, nullptr, w3, ss3,
                                featOut + (size_t)rowBase * C4, nullptr, lane);
    } else {
        run_layer<C3, C4, R, 2>(bufB, nullptr, w3, ss3,
                                nullptr, scalarOut + rowBase, lane);
    }
}

// ---------------------------------------------------------------------------
// cv2 assembly: K-max pool of patch feat + concat [cv1(6), dist, feat, patch]
// padded to 288 channels (f16).
// ---------------------------------------------------------------------------
__global__ __launch_bounds__(128) void cv2_kernel(
    const half_t* __restrict__ feat,   // [rows][128]
    const half_t* __restrict__ cv1h,   // [rows][32]
    const float* __restrict__ dist,    // [rows]
    half_t* __restrict__ cv2,          // [rows][288]
    int K)
{
    const int bm = blockIdx.x;
    const int c  = threadIdx.x;   // 0..127
    const size_t row0 = (size_t)bm * K;

    float mx = -3.0e38f;
    for (int k = 0; k < K; ++k)
        mx = fmaxf(mx, (float)feat[(row0 + k) * 128 + c]);
    half_t mh = (half_t)mx;

    for (int k = 0; k < K; ++k) {
        size_t r = row0 + k;
        half_t* dst = cv2 + r * 288;
        dst[7 + c]   = feat[r * 128 + c];
        dst[135 + c] = mh;
        if (c < 6)  dst[c] = cv1h[r * 32 + c];
        if (c == 6) dst[6] = (half_t)dist[r];
        if (c < 25) dst[263 + c] = (half_t)0.f;
    }
}

// ---------------------------------------------------------------------------
// Final: dual softmax over K, udf = sum w1*dist, grad = normalize(sum w2*on)
// ---------------------------------------------------------------------------
__global__ void final_kernel(
    const float* __restrict__ attL, const float* __restrict__ gattL,
    const float* __restrict__ dist, const float* __restrict__ onrm,
    float* __restrict__ out, int BM, int K)
{
    int t = blockIdx.x * blockDim.x + threadIdx.x;
    if (t >= BM) return;
    size_t r0 = (size_t)t * K;

    float m1 = -3.0e38f, m2 = -3.0e38f;
    for (int k = 0; k < K; ++k) {
        m1 = fmaxf(m1, attL[r0 + k]);
        m2 = fmaxf(m2, gattL[r0 + k]);
    }
    float e1[10], e2[10], s1 = 0.f, s2 = 0.f;
    for (int k = 0; k < K; ++k) {
        e1[k] = __expf(attL[r0 + k] - m1);  s1 += e1[k];
        e2[k] = __expf(gattL[r0 + k] - m2); s2 += e2[k];
    }
    float udf = 0.f, gx = 0.f, gy = 0.f, gz = 0.f;
    for (int k = 0; k < K; ++k) {
        float w1 = e1[k] / s1;
        float w2 = e2[k] / s2;
        udf += w1 * dist[r0 + k];
        gx += w2 * onrm[(r0 + k) * 3 + 0];
        gy += w2 * onrm[(r0 + k) * 3 + 1];
        gz += w2 * onrm[(r0 + k) * 3 + 2];
    }
    out[t] = udf;
    float nn = fmaxf(sqrtf(gx * gx + gy * gy + gz * gz), 1e-12f);
    out[BM + (size_t)t * 3 + 0] = gx / nn;
    out[BM + (size_t)t * 3 + 1] = gy / nn;
    out[BM + (size_t)t * 3 + 2] = gz / nn;
}

// ---------------------------------------------------------------------------
// Host: input layout assumes nested param dicts flatten in sorted-key order:
//   W0..W3, b0..b3, beta0..2, g0..2, mu0..2, var0..2  (20 arrays per net)
// ---------------------------------------------------------------------------
extern "C" void kernel_launch(void* const* d_in, const int* in_sizes, int n_in,
                              void* d_out, int out_size, void* d_ws, size_t ws_size,
                              hipStream_t stream) {
    (void)in_sizes; (void)n_in; (void)out_size; (void)ws_size;
    const int B = 2, N = 20000, M = 4096, Kk = 10;
    const int BM = B * M;
    const int rows = BM * Kk;   // 81920, multiple of 64

    const float* xyz   = (const float*)d_in[0];
    const float* nrm   = (const float*)d_in[1];
    const float* query = (const float*)d_in[2];

    char* ws = (char*)d_ws;
    size_t off = 0;
    auto take = [&](size_t bytes) -> void* {
        off = (off + 255) & ~(size_t)255;
        void* p = ws + off;
        off += bytes;
        return p;
    };

    int*    ws_idx  = (int*)   take((size_t)rows * 4);
    half_t* ws_cv1  = (half_t*)take((size_t)rows * 32 * 2);
    float*  ws_dist = (float*) take((size_t)rows * 4);
    float*  ws_on   = (float*) take((size_t)rows * 3 * 4);
    half_t* ws_feat = (half_t*)take((size_t)rows * 128 * 2);
    half_t* ws_cv2  = (half_t*)take((size_t)rows * 288 * 2);
    float*  ws_att  = (float*) take((size_t)rows * 4);
    float*  ws_gatt = (float*) take((size_t)rows * 4);

    // Padded (Kp = Cin pad32, Np = Cout pad16) layer dims
    static const int pKp[4] = {32, 64, 128, 128}, pNp[4] = {64, 128, 128, 128};
    static const int pCo[4] = {64, 128, 128, 128}, pCi[4] = {6, 64, 128, 128};
    static const int aKp[4] = {288, 256, 128, 32}, aNp[4] = {256, 128, 32, 16};
    static const int aCo[4] = {256, 128, 32, 1},   aCi[4] = {263, 256, 128, 32};

    half_t* wP[4]; float* sP[4];
    half_t* wA[4]; float* sA[4];
    half_t* wG[4]; float* sG[4];
    for (int l = 0; l < 4; ++l) { wP[l] = (half_t*)take((size_t)pKp[l] * pNp[l] * 2); sP[l] = (float*)take((size_t)pNp[l] * 8); }
    for (int l = 0; l < 4; ++l) { wA[l] = (half_t*)take((size_t)aKp[l] * aNp[l] * 2); sA[l] = (float*)take((size_t)aNp[l] * 8); }
    for (int l = 0; l < 4; ++l) { wG[l] = (half_t*)take((size_t)aKp[l] * aNp[l] * 2); sG[l] = (float*)take((size_t)aNp[l] * 8); }

    auto P = [&](int i) { return (const float*)d_in[i]; };
    const int PB = 3, AB = 23, GB = 43;  // sorted-key bases per net

    // Pack weights + fused scale/shift for all three nets
    for (int l = 0; l < 4; ++l) {
        {
            int T = (pNp[l] / 16) * (pKp[l] / 32) * 512;
            pack_wb<<<(T + 255) / 256, 256, 0, stream>>>(P(PB + l), wP[l], pCo[l], pCi[l], pKp[l], pNp[l]);
            const float* bb = P(PB + 4 + l);
            const float* gg = l < 3 ? P(PB + 11 + l) : bb;
            const float* be = l < 3 ? P(PB + 8 + l)  : bb;
            const float* mu = l < 3 ? P(PB + 14 + l) : bb;
            const float* va = l < 3 ? P(PB + 17 + l) : bb;
            pack_ss<<<1, pNp[l], 0, stream>>>(bb, gg, be, mu, va, sP[l], pCo[l], pNp[l], l == 3);
        }
        {
            int T = (aNp[l] / 16) * (aKp[l] / 32) * 512;
            pack_wb<<<(T + 255) / 256, 256, 0, stream>>>(P(AB + l), wA[l], aCo[l], aCi[l], aKp[l], aNp[l]);
            const float* bb = P(AB + 4 + l);
            const float* gg = l < 3 ? P(AB + 11 + l) : bb;
            const float* be = l < 3 ? P(AB + 8 + l)  : bb;
            const float* mu = l < 3 ? P(AB + 14 + l) : bb;
            const float* va = l < 3 ? P(AB + 17 + l) : bb;
            pack_ss<<<1, aNp[l], 0, stream>>>(bb, gg, be, mu, va, sA[l], aCo[l], aNp[l], l == 3);
        }
        {
            int T = (aNp[l] / 16) * (aKp[l] / 32) * 512;
            pack_wb<<<(T + 255) / 256, 256, 0, stream>>>(P(GB + l), wG[l], aCo[l], aCi[l], aKp[l], aNp[l]);
            const float* bb = P(GB + 4 + l);
            const float* gg = l < 3 ? P(GB + 11 + l) : bb;
            const float* be = l < 3 ? P(GB + 8 + l)  : bb;
            const float* mu = l < 3 ? P(GB + 14 + l) : bb;
            const float* va = l < 3 ? P(GB + 17 + l) : bb;
            pack_ss<<<1, aNp[l], 0, stream>>>(bb, gg, be, mu, va, sG[l], aCo[l], aNp[l], l == 3);
        }
    }

    // KNN + cv1 features
    knn_kernel<<<(BM + 7) / 8, 256, 0, stream>>>(xyz, query, ws_idx, B, N, M);
    cv1_kernel<<<(rows + 255) / 256, 256, 0, stream>>>(xyz, nrm, query, ws_idx,
                                                       ws_cv1, ws_dist, ws_on,
                                                       B, N, M, Kk);

    // Patch net (32->64->128->128->128): 1 wave/block, R=4 (64 rows), 32KB LDS
    {
        int blocks = rows / 64;
        size_t lds = (size_t)2 * 64 * 128 * sizeof(half_t);   // 32 KB
        mlp4_kernel<32, 64, 128, 128, 128, true, 4><<<blocks, 32, lds, stream>>>(
            ws_cv1, rows, wP[0], sP[0], wP[1], sP[1], wP[2], sP[2], wP[3], sP[3],
            ws_feat, nullptr);
    }

    // cv2 assembly (max-pool over K + concat)
    cv2_kernel<<<BM, 128, 0, stream>>>(ws_feat, ws_cv1, ws_dist, ws_cv2, Kk);

    // Attention nets (288->256->128->32->16): 1 wave/block, R=2 (32 rows), 36KB LDS
    {
        int blocks = rows / 32;
        size_t lds = (size_t)2 * 32 * 288 * sizeof(half_t);   // 36 KB
        mlp4_kernel<288, 256, 128, 32, 16, false, 2><<<blocks, 32, lds, stream>>>(
            ws_cv2, rows, wA[0], sA[0], wA[1], sA[1], wA[2], sA[2], wA[3], sA[3],
            nullptr, ws_att);
        mlp4_kernel<288, 256, 128, 32, 16, false, 2><<<blocks, 32, lds, stream>>>(
            ws_cv2, rows, wG[0], sG[0], wG[1], sG[1], wG[2], sG[2], wG[3], sG[3],
            nullptr, ws_gatt);
    }

    // Softmax + reductions + grad normalize
    final_kernel<<<(BM + 255) / 256, 256, 0, stream>>>(ws_att, ws_gatt, ws_dist,
                                                       ws_on, (float*)d_out, BM, Kk);
}
